// LSTMModel_71528385348058
// MI455X (gfx1250) — matrix operand order repaired and checked by reading, not compiled, not run on hardware
//
#include <hip/hip_runtime.h>
#include <hip/hip_bf16.h>
#include <math.h>

typedef __attribute__((ext_vector_type(16))) _Float16 v16h;
typedef __attribute__((ext_vector_type(8)))  float    v8f;

#define WMMA_F16(a, b, c) \
  __builtin_amdgcn_wmma_f32_16x16x32_f16(false, (a), false, (b), (short)0, (c), false, false)

__device__ __forceinline__ float sigm(float x) { return 1.0f / (1.0f + __expf(-x)); }

// ---------------------------------------------------------------------------
// one-shot f32 -> f16 weight conversion (removes v_cvt from all GEMM loops)
// ---------------------------------------------------------------------------
__global__ __launch_bounds__(256) void cvt_kernel(const float* __restrict__ in,
                                                  _Float16* __restrict__ out, int n) {
  int i = blockIdx.x * 256 + threadIdx.x;
  if (i < n) out[i] = (_Float16)in[i];
}

// ---------------------------------------------------------------------------
// conv1: implicit GEMM, M = 2048*400 = 819200, K = 4*8*8 = 256, N = 16.
// One wave per 16-row M tile; single N tile. Output [Bt,16,20,20] f16 + ReLU.
// ---------------------------------------------------------------------------
__global__ __launch_bounds__(256) void conv1_kernel(const float* __restrict__ obs,
                                                    const _Float16* __restrict__ w,   // [16][256] f16
                                                    const float* __restrict__ bias,
                                                    _Float16* __restrict__ out) {
  const int lane = threadIdx.x & 31;
  const int wave = threadIdx.x >> 5;
  const int tile = blockIdx.x * 8 + wave;            // 51200 tiles
  const int r    = lane & 15;
  const int hi   = (lane & 16) ? 1 : 0;
  const int kb_a = hi ? 8 : 0;
  const int kb_b = hi ? 16 : 0;

  const int grow_a = tile * 16 + r;
  const int bt_a   = grow_a / 400;
  const int pix_a  = grow_a % 400;
  const int oy = pix_a / 20, ox = pix_a % 20;
  const float* abase   = obs + (size_t)bt_a * 28224 + (oy * 4) * 84 + (ox * 4);
  const _Float16* wb   = w + (size_t)(lane & 15) * 256;

  v8f acc = {};
#pragma unroll
  for (int kk = 0; kk < 8; ++kk) {
    v16h a, b;
#pragma unroll
    for (int e = 0; e < 8; ++e) {
      int k0 = kk * 32 + kb_a + e;
      int k1 = k0 + 16;
      int c0 = k0 >> 6, rem0 = k0 & 63;
      int c1 = k1 >> 6, rem1 = k1 & 63;
      a[e]     = (_Float16)abase[c0 * 7056 + (rem0 >> 3) * 84 + (rem0 & 7)];
      a[e + 8] = (_Float16)abase[c1 * 7056 + (rem1 >> 3) * 84 + (rem1 & 7)];
    }
#pragma unroll
    for (int e = 0; e < 16; ++e) b[e] = wb[kk * 32 + kb_a + e + (e & 8)];
    acc = WMMA_F16(a, b, acc);
  }

  const int ch = lane & 15;
  const float bv = bias[ch];
#pragma unroll
  for (int j = 0; j < 8; ++j) {
    int m    = j + hi * 8;
    int grow = tile * 16 + m;
    int bt = grow / 400, pix = grow % 400;
    float v = acc[j] + bv;
    v = v > 0.0f ? v : 0.0f;
    out[(size_t)bt * 6400 + ch * 400 + pix] = (_Float16)v;
  }
}

// ---------------------------------------------------------------------------
// conv2: implicit GEMM, M = 2048*81 = 165888, K = 16*4*4 = 256, N = 32.
// Output flattened FC layout [Bt, 32*81] = [bt][ch*81 + pix], f16 + ReLU.
// ---------------------------------------------------------------------------
__global__ __launch_bounds__(256) void conv2_kernel(const _Float16* __restrict__ in, // [Bt][16][20][20]
                                                    const _Float16* __restrict__ w, // [32][256] f16
                                                    const float* __restrict__ bias,
                                                    _Float16* __restrict__ out) {   // [Bt][2592]
  const int lane = threadIdx.x & 31;
  const int wave = threadIdx.x >> 5;
  const int tile = blockIdx.x * 8 + wave;            // 10368 tiles
  const int r    = lane & 15;
  const int hi   = (lane & 16) ? 1 : 0;
  const int kb_a = hi ? 8 : 0;
  const int kb_b = hi ? 16 : 0;
  const int n    = lane & 15;

  const int grow_a = tile * 16 + r;
  const int bt_a   = grow_a / 81;
  const int pix_a  = grow_a % 81;
  const int oy = pix_a / 9, ox = pix_a % 9;
  const _Float16* abase = in + (size_t)bt_a * 6400 + (oy * 2) * 20 + (ox * 2);

  v8f acc0 = {}, acc1 = {};
#pragma unroll
  for (int kk = 0; kk < 8; ++kk) {
    v16h a, b0, b1;
#pragma unroll
    for (int e = 0; e < 8; ++e) {
      int k0 = kk * 32 + kb_a + e;
      int k1 = k0 + 16;
      int c0 = k0 >> 4, rem0 = k0 & 15;
      int c1 = k1 >> 4, rem1 = k1 & 15;
      a[e]     = abase[c0 * 400 + (rem0 >> 2) * 20 + (rem0 & 3)];
      a[e + 8] = abase[c1 * 400 + (rem1 >> 2) * 20 + (rem1 & 3)];
    }
    const _Float16* wb0 = w + (size_t)n * 256 + kk * 32 + kb_a;
    const _Float16* wb1 = w + (size_t)(16 + n) * 256 + kk * 32 + kb_a;
#pragma unroll
    for (int e = 0; e < 16; ++e) { b0[e] = wb0[e + (e & 8)]; b1[e] = wb1[e + (e & 8)]; }
    acc0 = WMMA_F16(a, b0, acc0);
    acc1 = WMMA_F16(a, b1, acc1);
  }

  const float bv0 = bias[n], bv1 = bias[16 + n];
#pragma unroll
  for (int j = 0; j < 8; ++j) {
    int m    = j + hi * 8;
    int grow = tile * 16 + m;
    int bt = grow / 81, pix = grow % 81;
    float v0 = acc0[j] + bv0; v0 = v0 > 0.0f ? v0 : 0.0f;
    float v1 = acc1[j] + bv1; v1 = v1 > 0.0f ? v1 : 0.0f;
    out[(size_t)bt * 2592 + n * 81 + pix]        = (_Float16)v0;
    out[(size_t)bt * 2592 + (16 + n) * 81 + pix] = (_Float16)v1;
  }
}

// ---------------------------------------------------------------------------
// fc: GEMM M=2048, K=2592, N=256, + bias + ReLU -> feat f16 [2048][256]
// ---------------------------------------------------------------------------
__global__ __launch_bounds__(256) void fc_kernel(const _Float16* __restrict__ in, // [2048][2592]
                                                 const _Float16* __restrict__ w, // [256][2592] f16
                                                 const float* __restrict__ bias,
                                                 _Float16* __restrict__ out) {   // [2048][256]
  const int lane  = threadIdx.x & 31;
  const int wave  = threadIdx.x >> 5;
  const int tile  = blockIdx.x * 8 + wave;           // 2048 tiles
  const int mtile = tile >> 4;
  const int ntile = tile & 15;
  const int r    = lane & 15;
  const int hi   = (lane & 16) ? 1 : 0;
  const int kb_a = hi ? 8 : 0;
  const int kb_b = hi ? 16 : 0;
  const int n    = lane & 15;

  const _Float16* abase = in + (size_t)(mtile * 16 + r) * 2592;
  const _Float16* wbase = w + (size_t)(ntile * 16 + n) * 2592;

  v8f acc = {};
  for (int kk = 0; kk < 81; ++kk) {              // K = 2592 = 81 * 32
    v16h a, b;
#pragma unroll
    for (int e = 0; e < 8; ++e) {
      a[e]     = abase[kk * 32 + kb_a + e];
      a[e + 8] = abase[kk * 32 + kb_a + 16 + e];
    }
#pragma unroll
    for (int e = 0; e < 16; ++e) b[e] = wbase[kk * 32 + kb_a + e + (e & 8)];
    acc = WMMA_F16(a, b, acc);
  }

  const int col = ntile * 16 + n;
  const float bv = bias[col];
#pragma unroll
  for (int j = 0; j < 8; ++j) {
    int m = j + hi * 8;
    float v = acc[j] + bv;
    v = v > 0.0f ? v : 0.0f;
    out[(size_t)(mtile * 16 + m) * 256 + col] = (_Float16)v;
  }
}

// ---------------------------------------------------------------------------
// gates_x = feat @ w_ih^T + (b_ih + b_hh): M=2048, K=256, N=1024 -> f32
// ---------------------------------------------------------------------------
__global__ __launch_bounds__(256) void gates_kernel(const _Float16* __restrict__ feat, // [2048][256]
                                                    const _Float16* __restrict__ wih, // [1024][256] f16
                                                    const float* __restrict__ bih,
                                                    const float* __restrict__ bhh,
                                                    float* __restrict__ gx) {         // [2048][1024]
  const int lane  = threadIdx.x & 31;
  const int wave  = threadIdx.x >> 5;
  const int tile  = blockIdx.x * 8 + wave;           // 8192 tiles
  const int mtile = tile >> 6;
  const int ntile = tile & 63;
  const int r    = lane & 15;
  const int hi   = (lane & 16) ? 1 : 0;
  const int kb_a = hi ? 8 : 0;
  const int kb_b = hi ? 16 : 0;
  const int n    = lane & 15;

  const _Float16* abase = feat + (size_t)(mtile * 16 + r) * 256;
  const _Float16* wbase = wih + (size_t)(ntile * 16 + n) * 256;

  v8f acc = {};
#pragma unroll
  for (int kk = 0; kk < 8; ++kk) {
    v16h a, b;
#pragma unroll
    for (int e = 0; e < 8; ++e) {
      a[e]     = abase[kk * 32 + kb_a + e];
      a[e + 8] = abase[kk * 32 + kb_a + 16 + e];
    }
#pragma unroll
    for (int e = 0; e < 16; ++e) b[e] = wbase[kk * 32 + kb_a + e + (e & 8)];
    acc = WMMA_F16(a, b, acc);
  }

  const int col = ntile * 16 + n;
  const float bv = bih[col] + bhh[col];
#pragma unroll
  for (int j = 0; j < 8; ++j) {
    int m = j + hi * 8;
    gx[(size_t)(mtile * 16 + m) * 1024 + col] = acc[j] + bv;
  }
}

// ---------------------------------------------------------------------------
// LSTM scan: 1 block, 16 waves. Wave w owns columns w*16..w*16+15 of each of
// the 4 gate quadrants -> i/f/g/o for a unit live in one lane's 4 accs.
// Per-step recurrent GEMM: M=16 (batch), K=256, N=1024 via WMMA from LDS h,
// with pre-converted f16 w_hh streamed from L2 (512 KB/step, no cvts).
// ---------------------------------------------------------------------------
__global__ __launch_bounds__(512) void lstm_kernel(const float* __restrict__ gx,      // [2048][1024]
                                                   const int* __restrict__ masks,     // [16][128]
                                                   const _Float16* __restrict__ whh,  // [1024][256] f16
                                                   const float* __restrict__ h0,      // [16][256]
                                                   const float* __restrict__ c0,      // [16][256]
                                                   float* __restrict__ feats,         // [2048][256]
                                                   float* __restrict__ dout) {
  __shared__ _Float16 sh[16 * 256];
  __shared__ float    sc[16 * 256];
  const int tid = threadIdx.x;

  for (int i = tid; i < 16 * 256; i += 512) {
    sh[i] = (_Float16)h0[i];
    sc[i] = c0[i];
  }
  __syncthreads();

  const int lane = tid & 31;
  const int wv   = tid >> 5;                   // 0..15
  const int r    = lane & 15;
  const int hi   = (lane & 16) ? 1 : 0;
  const int kb_a = hi ? 8 : 0;
  const int kb_b = hi ? 16 : 0;
  const int n    = lane & 15;
  const int unit = wv * 16 + n;                // hidden unit 0..255 owned by this lane

  for (int t = 0; t < 128; ++t) {
    // mask h,c at episode boundaries (before recurrent GEMM)
#pragma unroll
    for (int j = 0; j < 8; ++j) {
      int m = j + hi * 8;
      if (masks[m * 128 + t] == 0) {
        sh[m * 256 + unit] = (_Float16)0.0f;
        sc[m * 256 + unit] = 0.0f;
      }
    }
    __syncthreads();

    // g = gx[t] + h @ w_hh^T : wave computes its 16 columns of all 4 quadrants
    v8f acc[4] = {v8f{}, v8f{}, v8f{}, v8f{}};
    for (int kk = 0; kk < 8; ++kk) {
      v16h a;
      const _Float16* ab = sh + r * 256 + kk * 32;
#pragma unroll
      for (int e = 0; e < 8; ++e) { a[e] = ab[kb_a + e]; a[e + 8] = ab[kb_a + 16 + e]; }
#pragma unroll
      for (int q = 0; q < 4; ++q) {
        const _Float16* wb = whh + (size_t)(q * 256 + unit) * 256 + kk * 32 + kb_a;
        v16h b;
#pragma unroll
        for (int e = 0; e < 16; ++e) b[e] = wb[e + (e & 8)];
        acc[q] = WMMA_F16(a, b, acc[q]);
      }
    }
    __syncthreads();   // all reads of sh done before updates

    // elementwise cell update, fully in registers for owned (m, unit) pairs
#pragma unroll
    for (int j = 0; j < 8; ++j) {
      int m = j + hi * 8;
      const float* g = gx + (size_t)(m * 128 + t) * 1024;
      float gi = acc[0][j] + g[unit];
      float gf = acc[1][j] + g[256 + unit];
      float gg = acc[2][j] + g[512 + unit];
      float go = acc[3][j] + g[768 + unit];
      float cn = sigm(gf) * sc[m * 256 + unit] + sigm(gi) * tanhf(gg);
      float hn = sigm(go) * tanhf(cn);
      sc[m * 256 + unit] = cn;
      sh[m * 256 + unit] = (_Float16)hn;
      feats[(size_t)(m * 128 + t) * 256 + unit] = hn;
      if (t == 127) {
        dout[38912 + m * 256 + unit] = hn;   // hT [16,1,256]
        dout[43008 + m * 256 + unit] = cn;   // cT [16,1,256]
      }
    }
    __syncthreads();
  }
}

// ---------------------------------------------------------------------------
// heads: 2048 rows x (18 policy + 1 critic) dot products of length 256.
// ---------------------------------------------------------------------------
__global__ __launch_bounds__(256) void heads_kernel(const float* __restrict__ feats, // [2048][256]
                                                    const float* __restrict__ pw,   // [18][256]
                                                    const float* __restrict__ pb,
                                                    const float* __restrict__ cw,   // [1][256]
                                                    const float* __restrict__ cb,
                                                    float* __restrict__ dout) {
  int id = blockIdx.x * 256 + threadIdx.x;
  if (id >= 2048 * 19) return;
  int row = id / 19, o = id % 19;
  const float* f = feats + (size_t)row * 256;
  const float* w = (o < 18) ? (pw + (size_t)o * 256) : cw;
  float s = (o < 18) ? pb[o] : cb[0];
#pragma unroll 4
  for (int k = 0; k < 256; ++k) s += f[k] * w[k];
  if (o < 18) dout[row * 18 + o] = s;
  else        dout[36864 + row] = s;
}

// ---------------------------------------------------------------------------
extern "C" void kernel_launch(void* const* d_in, const int* in_sizes, int n_in,
                              void* d_out, int out_size, void* d_ws, size_t ws_size,
                              hipStream_t stream) {
  (void)in_sizes; (void)n_in; (void)out_size; (void)ws_size;
  const float* obs      = (const float*)d_in[0];
  const int*   masks    = (const int*)d_in[1];
  const float* h0       = (const float*)d_in[2];
  const float* c0       = (const float*)d_in[3];
  const float* conv1_w  = (const float*)d_in[4];
  const float* conv1_b  = (const float*)d_in[5];
  const float* conv2_w  = (const float*)d_in[6];
  const float* conv2_b  = (const float*)d_in[7];
  const float* fc_w     = (const float*)d_in[8];
  const float* fc_b     = (const float*)d_in[9];
  const float* w_ih     = (const float*)d_in[10];
  const float* w_hh     = (const float*)d_in[11];
  const float* b_ih     = (const float*)d_in[12];
  const float* b_hh     = (const float*)d_in[13];
  const float* policy_w = (const float*)d_in[14];
  const float* policy_b = (const float*)d_in[15];
  const float* critic_w = (const float*)d_in[16];
  const float* critic_b = (const float*)d_in[17];
  float* out = (float*)d_out;

  char* ws = (char*)d_ws;
  _Float16* conv1o = (_Float16*)(ws + 0);         // 2048*6400 f16 = 26,214,400 B
  _Float16* conv2o = (_Float16*)(ws + 26214400);  // 2048*2592 f16 = 10,616,832 B
  _Float16* feat16 = (_Float16*)(ws + 36831232);  // 2048*256  f16 =  1,048,576 B
  float*    gxbuf  = (float*)   (ws + 37879808);  // 2048*1024 f32 =  8,388,608 B
  float*    featsb = (float*)   (ws + 46268416);  // 2048*256  f32 =  2,097,152 B
  _Float16* whh16  = (_Float16*)(ws + 48365568);  // 1024*256  f16 =    524,288 B
  _Float16* wih16  = (_Float16*)(ws + 48889856);  // 1024*256  f16 =    524,288 B
  _Float16* fcw16  = (_Float16*)(ws + 49414144);  // 256*2592  f16 =  1,327,104 B
  _Float16* c1w16  = (_Float16*)(ws + 50741248);  // 16*256    f16 =      8,192 B
  _Float16* c2w16  = (_Float16*)(ws + 50749440);  // 32*256    f16 =     16,384 B

  // one-shot weight conversions (tiny; off the GEMM hot loops)
  cvt_kernel<<<(262144 + 255) / 256, 256, 0, stream>>>(w_hh, whh16, 262144);
  cvt_kernel<<<(262144 + 255) / 256, 256, 0, stream>>>(w_ih, wih16, 262144);
  cvt_kernel<<<(663552 + 255) / 256, 256, 0, stream>>>(fc_w, fcw16, 663552);
  cvt_kernel<<<(4096 + 255) / 256, 256, 0, stream>>>(conv1_w, c1w16, 4096);
  cvt_kernel<<<(8192 + 255) / 256, 256, 0, stream>>>(conv2_w, c2w16, 8192);

  conv1_kernel<<<6400, 256, 0, stream>>>(obs, c1w16, conv1_b, conv1o);
  conv2_kernel<<<1296, 256, 0, stream>>>(conv1o, c2w16, conv2_b, conv2o);
  fc_kernel<<<256, 256, 0, stream>>>(conv2o, fcw16, fc_b, feat16);
  gates_kernel<<<1024, 256, 0, stream>>>(feat16, wih16, b_ih, b_hh, gxbuf);
  lstm_kernel<<<1, 512, 0, stream>>>(gxbuf, masks, whh16, h0, c0, featsb, out);
  heads_kernel<<<152, 256, 0, stream>>>(featsb, policy_w, policy_b, critic_w, critic_b, out);
}
